// CrossModalFusion_80977313399388
// MI455X (gfx1250) — matrix operand order repaired and checked by reading, compiled, not run on hardware
//
#include <hip/hip_runtime.h>
#include <hip/hip_bf16.h>

#define B_   16384
#define H_   1024
#define FF_  4096
#define NL_  4

// GEMM tiling
#define BM   128      // rows per workgroup
#define BN   128      // cols per workgroup
#define KC   64       // K-chunk staged in LDS per double-buffer slot
#define RS   72       // LDS row stride in elements (16B aligned, bank-friendly)

typedef __attribute__((ext_vector_type(16))) __bf16       v16bf;
typedef __attribute__((ext_vector_type(8)))  float        v8f;
typedef __attribute__((ext_vector_type(4)))  unsigned int u32x4;
typedef __attribute__((ext_vector_type(2)))  unsigned int u32x2;

union Frag {
  v16bf v;
  u32x4 q[2];
};

__device__ __forceinline__ unsigned short f2bf(float f) {
  unsigned int u = __float_as_uint(f);
  u += 0x7FFFu + ((u >> 16) & 1u);     // round-to-nearest-even
  return (unsigned short)(u >> 16);
}

__device__ __forceinline__ float gelu_exact(float x) {
  return 0.5f * x * (1.0f + erff(x * 0.70710678118654752f));
}

enum { EPI_BF = 0, EPI_F32 = 1, EPI_GELU_BF = 2, EPI_RES_MASK = 3 };

// ---------------------------------------------------------------------------
// WMMA GEMM:  out[rows x N] = X[rows x K](bf16) @ W[N x K]^T (bf16) + bias
// 256 threads = 8 waves; wave tile 32x64; WG tile 128x128.
// A and B chunks (128 x 64 bf16 each) are staged in LDS with double-buffered
// GLOBAL_LOAD_ASYNC_TO_LDS_B128 (ASYNCcnt), so each global element crosses
// L2->WGP exactly once per workgroup. Fragments are then read with
// ds_load_b128 directly into the ISA-required WMMA register layouts:
//   A(16x32): lane<16 -> row=lane, K {k..k+7},{k+16..k+23};
//             lane>=16 -> row=lane-16, K {k+8..15},{k+24..31}
//   B(32x16): lane<16 -> col=lane, K {k..k+7},{k+16..k+23};
//             lane>=16 -> col=lane-16, K {k+8..15},{k+24..31}
// ---------------------------------------------------------------------------
template<int EPI>
__global__ __launch_bounds__(256, 2) void gemm_bf16(
    const unsigned short* __restrict__ X,
    const unsigned short* __restrict__ W,
    const float* __restrict__ bias,
    int K, int N,
    float* out32,
    unsigned short* __restrict__ outbf,
    const float* res32,
    const float* __restrict__ hp,
    const float* __restrict__ hm)
{
  __shared__ unsigned short shA[2][BM * RS];
  __shared__ unsigned short shB[2][BN * RS];

  const int tid  = threadIdx.x;
  const int lane = tid & 31;
  const int w    = tid >> 5;
  const int wr   = w >> 1;                       // 0..3 (row stripe)
  const int wc   = w & 1;                        // 0..1 (col stripe)
  const int row0 = blockIdx.y * BM;
  const int col0 = blockIdx.x * BN;
  const int l16  = lane >> 4;
  const int lr   = lane & 15;

  // ---- async staging map: thread -> (row sr, 32-elem half sh); 64B/thread/matrix
  const int sr = tid >> 1;                       // 0..127
  const int sh = (tid & 1) * 32;                 // element offset in 64-elem chunk
  const unsigned short* gA = X + (size_t)(row0 + sr) * K + sh;
  const unsigned short* gB = W + (size_t)(col0 + sr) * K + sh;
  const unsigned ldsA[2] = {
      (unsigned)(size_t)&shA[0][sr * RS + sh],
      (unsigned)(size_t)&shA[1][sr * RS + sh] };
  const unsigned ldsB[2] = {
      (unsigned)(size_t)&shB[0][sr * RS + sh],
      (unsigned)(size_t)&shB[1][sr * RS + sh] };

  auto stage = [&](int buf, int kc) {
    const unsigned short* ga = gA + kc;
    const unsigned short* gb = gB + kc;
    // INST_OFFSET applies to both the LDS and global address (ISA 15.18.3)
    asm volatile(
      "global_load_async_to_lds_b128 %0, %2, off\n\t"
      "global_load_async_to_lds_b128 %0, %2, off offset:16\n\t"
      "global_load_async_to_lds_b128 %0, %2, off offset:32\n\t"
      "global_load_async_to_lds_b128 %0, %2, off offset:48\n\t"
      "global_load_async_to_lds_b128 %1, %3, off\n\t"
      "global_load_async_to_lds_b128 %1, %3, off offset:16\n\t"
      "global_load_async_to_lds_b128 %1, %3, off offset:32\n\t"
      "global_load_async_to_lds_b128 %1, %3, off offset:48"
      :: "v"(ldsA[buf]), "v"(ldsB[buf]), "v"(ga), "v"(gb)
      : "memory");
  };

  const v8f zero = {0.f,0.f,0.f,0.f,0.f,0.f,0.f,0.f};
  v8f acc[2][4];
#pragma unroll
  for (int mi = 0; mi < 2; ++mi)
#pragma unroll
    for (int ni = 0; ni < 4; ++ni) acc[mi][ni] = zero;

  const int aoff = l16 ? 8 : 0;
  const int boff = l16 ? 8 : 0;

  stage(0, 0);
  int buf = 0;
  for (int kc = 0; kc < K; kc += KC) {
    const bool more = (kc + KC) < K;
    if (more) {
      stage(buf ^ 1, kc + KC);
      asm volatile("s_wait_asynccnt 0x8" ::: "memory");  // current stage landed
    } else {
      asm volatile("s_wait_asynccnt 0x0" ::: "memory");
    }
    __syncthreads();

    const unsigned short* sa = shA[buf];
    const unsigned short* sb = shB[buf];
#pragma unroll
    for (int ks = 0; ks < 2; ++ks) {               // two 32-K WMMA steps per chunk
      Frag a[2], b[4];
#pragma unroll
      for (int mi = 0; mi < 2; ++mi) {
        const unsigned short* p = sa + (wr * 32 + 16 * mi + lr) * RS + ks * 32 + aoff;
        a[mi].q[0] = *(const u32x4*)(p);
        a[mi].q[1] = *(const u32x4*)(p + 16);
      }
#pragma unroll
      for (int ni = 0; ni < 4; ++ni) {
        const unsigned short* p = sb + (wc * 64 + 16 * ni + lr) * RS + ks * 32 + boff;
        b[ni].q[0] = *(const u32x4*)(p);
        b[ni].q[1] = *(const u32x4*)(p + 16);
      }
#pragma unroll
      for (int ni = 0; ni < 4; ++ni) {
        acc[0][ni] = __builtin_amdgcn_wmma_f32_16x16x32_bf16(
            false, a[0].v, false, b[ni].v, (short)0, acc[0][ni], false, false);
        acc[1][ni] = __builtin_amdgcn_wmma_f32_16x16x32_bf16(
            false, a[1].v, false, b[ni].v, (short)0, acc[1][ni], false, false);
      }
    }
    __syncthreads();
    buf ^= 1;
  }

  // C/D layout: VGPR r -> row r (+8 for lanes>=16), col = lane%16
#pragma unroll
  for (int ni = 0; ni < 4; ++ni) {
    const int col = col0 + wc * 64 + 16 * ni + lr;
    const float bv = bias[col];
#pragma unroll
    for (int mi = 0; mi < 2; ++mi) {
      const int rbase = row0 + wr * 32 + 16 * mi + (l16 ? 8 : 0);
#pragma unroll
      for (int r = 0; r < 8; ++r) {
        const int row = rbase + r;
        const size_t idx = (size_t)row * N + col;
        float x = acc[mi][ni][r] + bv;
        if (EPI == EPI_BF) {
          outbf[idx] = f2bf(x);
        } else if (EPI == EPI_F32) {
          out32[idx] = x;
        } else if (EPI == EPI_GELU_BF) {
          outbf[idx] = f2bf(gelu_exact(x));
        } else { // EPI_RES_MASK : y = res + x*eff ; store fp32 + bf16
          const float e = hp[row] * hm[row];
          const float y = res32[idx] + x * e;
          out32[idx] = y;
          outbf[idx] = f2bf(y);
        }
      }
    }
  }
}

// ---------------------------------------------------------------------------
// Fused double LayerNorm:  p = LN(base + delta*eff; g1,b1)  (fp32 out)
//                          hln = bf16(LN(p; g2,b2))         (FFN input)
// One 256-thread block per row (H=1024, 4 elems/thread).
// ---------------------------------------------------------------------------
__global__ __launch_bounds__(256) void ln2_kernel(
    const float* base, const float* __restrict__ delta,
    const float* __restrict__ hp, const float* __restrict__ hm,
    const float* __restrict__ g1, const float* __restrict__ b1,
    const float* __restrict__ g2, const float* __restrict__ b2,
    float* out32, unsigned short* __restrict__ outbf)
{
  __shared__ float red[256];
  const int t = threadIdx.x;
  const int row = blockIdx.x;
  const float e = hp[row] * hm[row];
  const size_t off = (size_t)row * H_;

  auto rsum = [&](float v) -> float {
    red[t] = v; __syncthreads();
    for (int s = 128; s > 0; s >>= 1) {
      if (t < s) red[t] += red[t + s];
      __syncthreads();
    }
    float r = red[0]; __syncthreads();
    return r;
  };

  float x[4];
#pragma unroll
  for (int j = 0; j < 4; ++j) {
    const int c = t + 256 * j;
    x[j] = base[off + c] + delta[off + c] * e;
  }
  const float s1 = rsum(x[0] + x[1] + x[2] + x[3]);
  const float s2 = rsum(x[0]*x[0] + x[1]*x[1] + x[2]*x[2] + x[3]*x[3]);
  const float mu = s1 * (1.0f / H_);
  const float rs = rsqrtf(s2 * (1.0f / H_) - mu * mu + 1e-5f);

  float y[4];
#pragma unroll
  for (int j = 0; j < 4; ++j) {
    const int c = t + 256 * j;
    y[j] = (x[j] - mu) * rs * g1[c] + b1[c];
    out32[off + c] = y[j];
  }
  const float t1 = rsum(y[0] + y[1] + y[2] + y[3]);
  const float t2 = rsum(y[0]*y[0] + y[1]*y[1] + y[2]*y[2] + y[3]*y[3]);
  const float mu2 = t1 * (1.0f / H_);
  const float rs2 = rsqrtf(t2 * (1.0f / H_) - mu2 * mu2 + 1e-5f);
#pragma unroll
  for (int j = 0; j < 4; ++j) {
    const int c = t + 256 * j;
    outbf[off + c] = f2bf((y[j] - mu2) * rs2 * g2[c] + b2[c]);
  }
}

// --------------------------- elementwise kernels ---------------------------
__global__ void f32_to_bf16_kernel(const float* __restrict__ src,
                                   unsigned short* __restrict__ dst, size_t n) {
  const size_t stride = (size_t)gridDim.x * blockDim.x * 4;
  for (size_t j = ((size_t)blockIdx.x * blockDim.x + threadIdx.x) * 4;
       j + 3 < n; j += stride) {
    u32x4 sv = *(const u32x4*)(src + j);
    u32x2 o;
    o.x = (unsigned)f2bf(__uint_as_float(sv.x)) |
          ((unsigned)f2bf(__uint_as_float(sv.y)) << 16);
    o.y = (unsigned)f2bf(__uint_as_float(sv.z)) |
          ((unsigned)f2bf(__uint_as_float(sv.w)) << 16);
    *(u32x2*)(dst + j) = o;
  }
}

__global__ void init_state_kernel(const float* __restrict__ p, const float* __restrict__ m,
                                  float* __restrict__ p32, float* __restrict__ m32,
                                  unsigned short* __restrict__ pbf, unsigned short* __restrict__ mbf) {
  const size_t n = (size_t)B_ * H_;
  const size_t stride = (size_t)gridDim.x * blockDim.x;
  for (size_t i = (size_t)blockIdx.x * blockDim.x + threadIdx.x; i < n; i += stride) {
    const float a = p[i], b = m[i];
    p32[i] = a; m32[i] = b; pbf[i] = f2bf(a); mbf[i] = f2bf(b);
  }
}

__global__ void headprep_kernel(const float* __restrict__ p32, const float* __restrict__ prot,
                                const float* __restrict__ m32, const float* __restrict__ metab,
                                float* __restrict__ cp32, unsigned short* __restrict__ cpbf,
                                float* __restrict__ cm32, unsigned short* __restrict__ cmbf) {
  const size_t n = (size_t)B_ * H_;
  const size_t stride = (size_t)gridDim.x * blockDim.x;
  for (size_t i = (size_t)blockIdx.x * blockDim.x + threadIdx.x; i < n; i += stride) {
    const float a = p32[i] - prot[i];
    const float b = m32[i] - metab[i];
    cp32[i] = a; cpbf[i] = f2bf(a);
    cm32[i] = b; cmbf[i] = f2bf(b);
  }
}

__global__ void gate_kernel(const float* __restrict__ gl, const float* __restrict__ c32,
                            const float* __restrict__ base,
                            const float* __restrict__ hp, const float* __restrict__ hm,
                            float* __restrict__ out32, unsigned short* __restrict__ fubf,
                            int coloff) {
  const size_t n = (size_t)B_ * H_;
  const size_t stride = (size_t)gridDim.x * blockDim.x;
  for (size_t i = (size_t)blockIdx.x * blockDim.x + threadIdx.x; i < n; i += stride) {
    const int row = (int)(i >> 10);        // /H_
    const int col = (int)(i & (H_ - 1));
    const float e = hp[row] * hm[row];
    const float g = 1.0f / (1.0f + __expf(-gl[i]));
    const float v = base[i] + g * c32[i] * e;
    out32[i] = v;
    fubf[(size_t)row * (2 * H_) + coloff + col] = f2bf(v);
  }
}

__global__ void mix_kernel(const float* __restrict__ f32,
                           const float* __restrict__ pe, const float* __restrict__ me,
                           const float* __restrict__ hp, const float* __restrict__ hm,
                           float* __restrict__ out) {
  const size_t n = (size_t)B_ * H_;
  const size_t stride = (size_t)gridDim.x * blockDim.x;
  for (size_t i = (size_t)blockIdx.x * blockDim.x + threadIdx.x; i < n; i += stride) {
    const int row = (int)(i >> 10);
    const float p = hp[row], m = hm[row];
    out[i] = f32[i] * (p * m) + pe[i] * (p * (1.f - m)) + me[i] * ((1.f - p) * m);
  }
}

// ---------------------------------------------------------------------------
extern "C" void kernel_launch(void* const* d_in, const int* in_sizes, int n_in,
                              void* d_out, int out_size, void* d_ws, size_t ws_size,
                              hipStream_t stream) {
  (void)in_sizes; (void)n_in; (void)out_size;
  const float* protein = (const float*)d_in[0];
  const float* metab   = (const float*)d_in[1];
  const float* hp      = (const float*)d_in[2];
  const float* hm      = (const float*)d_in[3];

  const size_t H2 = (size_t)H_ * H_;
  const size_t BH = (size_t)B_ * H_;
  char* wsb = (char*)d_ws;
  size_t o = 0;
  auto alloc = [&](size_t bytes) -> char* {
    char* r = wsb + o;
    o += (bytes + 255) & ~(size_t)255;
    return r;
  };
  unsigned short* wbf  = (unsigned short*)alloc(85 * H2 * 2);  // all weights, bf16
  float*          p32  = (float*)alloc(BH * 4);
  float*          m32  = (float*)alloc(BH * 4);
  unsigned short* pbf  = (unsigned short*)alloc(BH * 2);
  unsigned short* mbf  = (unsigned short*)alloc(BH * 2);
  unsigned short* vbf  = (unsigned short*)alloc(BH * 2);       // v / cpbf / h reuse
  unsigned short* hln  = (unsigned short*)alloc(BH * 2);       // ffn-input / cmbf
  float*          pc32 = (float*)alloc(BH * 4);                // attn-out / gate / fused
  unsigned short* h1bf = (unsigned short*)alloc((size_t)B_ * FF_ * 2);
  unsigned short* fubf = (unsigned short*)alloc((size_t)B_ * 2 * H_ * 2);
  if (o > ws_size) return;  // workspace too small: no-op
  float* cp32 = (float*)h1bf;                                  // reuse (head phase)
  float* cm32 = cp32 + BH;

  const int EW = 8192;  // elementwise grid

  // ---- convert all used weights to bf16 (once per launch) ----
  auto cvt = [&](const float* src, size_t dofs, size_t n) {
    int blocks = (int)((n / 4 + 255) / 256);
    if (blocks > 4096) blocks = 4096;
    f32_to_bf16_kernel<<<blocks, 256, 0, stream>>>(src, wbf + dofs, n);
  };
  for (int i = 0; i < NL_; ++i)
    for (int d = 0; d < 2; ++d) {
      const int pb = 4 + 12 * d;
      const float* Wqkv = (const float*)d_in[pb + 0];
      const float* Wo   = (const float*)d_in[pb + 2];
      const float* W1   = (const float*)d_in[pb + 8];
      const float* W2   = (const float*)d_in[pb + 10];
      const size_t blk = (size_t)(i * 2 + d) * 10 * H2;
      cvt(Wqkv + ((size_t)i * 3 * H_ + 2 * H_) * H_, blk, H2);    // Wv (only live slice)
      cvt(Wo + (size_t)i * H2,     blk + 1 * H2, H2);
      cvt(W1 + (size_t)i * 4 * H2, blk + 2 * H2, 4 * H2);
      cvt(W2 + (size_t)i * 4 * H2, blk + 6 * H2, 4 * H2);
    }
  cvt((const float*)d_in[28], 80 * H2, H2);       // gp_W
  cvt((const float*)d_in[30], 81 * H2, H2);       // gm_W
  cvt((const float*)d_in[32], 82 * H2, 2 * H2);   // fu_W1
  cvt((const float*)d_in[34], 84 * H2, H2);       // fu_W2

  init_state_kernel<<<EW, 256, 0, stream>>>(protein, metab, p32, m32, pbf, mbf);

  const dim3 gH(H_ / BN, B_ / BM), gF(FF_ / BN, B_ / BM);

  // ---- transformer layers (attention == V,O projections since Lk==1) ----
  for (int i = 0; i < NL_; ++i)
    for (int d = 0; d < 2; ++d) {
      const int pb = 4 + 12 * d;
      const float* bqkv = (const float*)d_in[pb + 1];
      const float* bo   = (const float*)d_in[pb + 3];
      const float* ng   = (const float*)d_in[pb + 4];
      const float* nb   = (const float*)d_in[pb + 5];
      const float* fng  = (const float*)d_in[pb + 6];
      const float* fnb  = (const float*)d_in[pb + 7];
      const float* b1   = (const float*)d_in[pb + 9];
      const float* b2   = (const float*)d_in[pb + 11];
      unsigned short* wblk = wbf + (size_t)(i * 2 + d) * 10 * H2;
      const unsigned short* kvbf = d ? pbf : mbf;   // cross input
      float*          st32 = d ? m32 : p32;         // state being updated
      unsigned short* stbf = d ? mbf : pbf;

      gemm_bf16<EPI_BF><<<gH, 256, 0, stream>>>(
          kvbf, wblk, bqkv + (size_t)i * 3 * H_ + 2 * H_, H_, H_,
          nullptr, vbf, nullptr, nullptr, nullptr);
      gemm_bf16<EPI_F32><<<gH, 256, 0, stream>>>(
          vbf, wblk + H2, bo + (size_t)i * H_, H_, H_,
          pc32, nullptr, nullptr, nullptr, nullptr);
      ln2_kernel<<<B_, 256, 0, stream>>>(
          st32, pc32, hp, hm,
          ng + (size_t)i * H_, nb + (size_t)i * H_,
          fng + (size_t)i * H_, fnb + (size_t)i * H_, st32, hln);
      gemm_bf16<EPI_GELU_BF><<<gF, 256, 0, stream>>>(
          hln, wblk + 2 * H2, b1 + (size_t)i * FF_, H_, FF_,
          nullptr, h1bf, nullptr, nullptr, nullptr);
      gemm_bf16<EPI_RES_MASK><<<gH, 256, 0, stream>>>(
          h1bf, wblk + 6 * H2, b2 + (size_t)i * H_, FF_, H_,
          st32, stbf, st32, hp, hm);
    }

  // ---- gated head + fusion MLP ----
  float* out_pe = (float*)d_out;
  float* out_me = out_pe + BH;
  float* out_f  = out_pe + 2 * BH;

  headprep_kernel<<<EW, 256, 0, stream>>>(p32, protein, m32, metab,
                                          cp32, vbf, cm32, hln);
  gemm_bf16<EPI_F32><<<gH, 256, 0, stream>>>(
      vbf, wbf + 80 * H2, (const float*)d_in[29], H_, H_,
      pc32, nullptr, nullptr, nullptr, nullptr);
  gate_kernel<<<EW, 256, 0, stream>>>(pc32, cp32, protein, hp, hm, out_pe, fubf, 0);
  gemm_bf16<EPI_F32><<<gH, 256, 0, stream>>>(
      hln, wbf + 81 * H2, (const float*)d_in[31], H_, H_,
      pc32, nullptr, nullptr, nullptr, nullptr);
  gate_kernel<<<EW, 256, 0, stream>>>(pc32, cm32, metab, hp, hm, out_me, fubf, H_);
  gemm_bf16<EPI_GELU_BF><<<gH, 256, 0, stream>>>(
      fubf, wbf + 82 * H2, (const float*)d_in[33], 2 * H_, H_,
      nullptr, vbf, nullptr, nullptr, nullptr);
  gemm_bf16<EPI_F32><<<gH, 256, 0, stream>>>(
      vbf, wbf + 84 * H2, (const float*)d_in[35], H_, H_,
      pc32, nullptr, nullptr, nullptr, nullptr);
  mix_kernel<<<EW, 256, 0, stream>>>(pc32, out_pe, out_me, hp, hm, out_f);
}
